// DSEdgeReadout_10582799417476
// MI455X (gfx1250) — matrix operand order repaired and checked by reading, not compiled, not run on hardware
//
#include <hip/hip_runtime.h>
#include <hip/hip_bf16.h>

// ---------------------------------------------------------------------------
// DSEdgeReadout for MI455X (gfx1250, wave32, WMMA)
//   h = relu(edge_attr @ W1 + b1)            [E,128] bf16 WMMA, f32 accum
//   scatter-mean over batch[edge_index[0]]   LDS ds_add_f32 + global atomics
//   out = relu(feat @ W2 + b2) @ W3 + b3     tiny, plain f32
// ---------------------------------------------------------------------------

typedef __attribute__((ext_vector_type(16))) __bf16 v16bf;
typedef __attribute__((ext_vector_type(8)))  __bf16 v8bf;
typedef __attribute__((ext_vector_type(8)))  float  v8f;

#define GRAPHS   64
#define HDIM     128
#define INDIM    128
#define OUTDIM   64
#define NTHREADS 256          // 8 waves of 32
#define EPB      1024         // edges per block
#define TILES_PER_WAVE 8      // EPB / (8 waves * 16 edges)
#define PAIRS_PER_WAVE (TILES_PER_WAVE / 2)

// LDS bank-conflict-free strides (64 banks x 4B):
//  - W1^T rows: 136 bf16 = 68 dwords -> B-frag start bank = (col*4+..)%64,
//    16 lanes x b128 tile all 64 banks exactly once (was: stride 64 dwords
//    -> all 16 lanes on one 4-bank group, 16-way serialization).
//  - ssum rows: 132 f32 -> scatter bank = (g*4 + col)%64, spread by graph id
//    (was: stride 128 -> lanes 0-15/16-31 collide on the same 16 banks).
#define W1T_STRIDE  136
#define SSUM_STRIDE 132

// Native RNE f32 -> bf16 conversion; backend packs pairs into v_cvt_pk_bf16_f32.
static __device__ __forceinline__ __bf16 f2bf(float f) {
    return (__bf16)f;
}

static __device__ __forceinline__ float relu1(float x) {
    return __builtin_amdgcn_fmed3f(x, 0.0f, __builtin_inff());
}

// ---------------------------------------------------------------------------
__global__ __launch_bounds__(NTHREADS)
void zero_ws_kernel(float* ws, int n) {
    for (int i = threadIdx.x; i < n; i += NTHREADS) ws[i] = 0.0f;
}

// ---------------------------------------------------------------------------
// Kernel 1: per-edge Linear+ReLU with WMMA, scatter-add into per-block LDS
// accumulators (row GRAPHS is a dummy row for out-of-range tail edges so the
// scatter is branch-free), then one global-atomic flush per block.
//
// Each wave processes PAIRS of 16-edge tiles so every B fragment loaded from
// LDS feeds two independent WMMA chains: halves B-side LDS traffic and removes
// D->C dependency stalls between consecutive WMMAs.
// waves_per_eu(4) caps VGPRs at 256 (no s_set_vgpr_msb, 4 waves/SIMD).
// ---------------------------------------------------------------------------
__global__ __launch_bounds__(NTHREADS)
__attribute__((amdgpu_waves_per_eu(4)))
void edge_mlp_scatter_kernel(const int*   __restrict__ edge_src,   // [E] (row 0 of edge_index)
                             const float* __restrict__ edge_attr,  // [E][128]
                             const int*   __restrict__ batch,      // [N]
                             const float* __restrict__ W1,         // [128][128]
                             const float* __restrict__ b1,         // [128]
                             float*       __restrict__ gsums,      // [64][128]
                             float*       __restrict__ gcounts,    // [64]
                             int E) {
    __shared__ __align__(16) __bf16 w1t[HDIM * W1T_STRIDE];   // W1^T padded, ~34 KB
    __shared__ float ssum[(GRAPHS + 1) * SSUM_STRIDE];        // padded + dummy row
    __shared__ float scnt[GRAPHS];

    const int tid   = threadIdx.x;
    const int wave  = tid >> 5;
    const int lane  = tid & 31;
    const int lhalf = lane >> 4;   // 0: lanes 0-15, 1: lanes 16-31
    const int l16   = lane & 15;

    // Stage W1^T as bf16 in LDS (coalesced global read, scattered LDS write).
    for (int i = tid; i < INDIM * HDIM; i += NTHREADS) {
        const int k = i >> 7;      // input dim
        const int h = i & 127;     // output dim
        w1t[h * W1T_STRIDE + k] = f2bf(W1[i]);
    }
    for (int i = tid; i < (GRAPHS + 1) * SSUM_STRIDE; i += NTHREADS) ssum[i] = 0.0f;
    if (tid < GRAPHS) scnt[tid] = 0.0f;
    __syncthreads();

    // Hoist the 8 per-lane biases (b1[nt*16 + lane%16]) out of the hot loop.
    float bias[8];
    #pragma unroll
    for (int nt = 0; nt < 8; ++nt) bias[nt] = b1[nt * 16 + l16];

    const int blockBase = blockIdx.x * EPB;

    // Loader for one 16x128 A-tile (4 K-step fragments) starting at `row`,
    // per documented 16-bit A layout: lane holds row M = lane%16; low lane
    // half K {b..b+7, b+16..b+23}, high half +8. f32 -> bf16 on the fly.
    auto load_afrags = [&](int row, v16bf a4[4]) {
        const bool rv = (row < E);
        const float* ap = edge_attr + (size_t)row * INDIM + lhalf * 8;
        #pragma unroll
        for (int kk = 0; kk < 4; ++kk) {
            v16bf a;
            #pragma unroll
            for (int i = 0; i < 16; ++i) a[i] = f2bf(0.0f);
            if (rv) {
                const float4 x0 = *reinterpret_cast<const float4*>(ap + kk * 32 + 0);
                const float4 x1 = *reinterpret_cast<const float4*>(ap + kk * 32 + 4);
                const float4 y0 = *reinterpret_cast<const float4*>(ap + kk * 32 + 16);
                const float4 y1 = *reinterpret_cast<const float4*>(ap + kk * 32 + 20);
                a[0]  = f2bf(x0.x); a[1]  = f2bf(x0.y); a[2]  = f2bf(x0.z); a[3]  = f2bf(x0.w);
                a[4]  = f2bf(x1.x); a[5]  = f2bf(x1.y); a[6]  = f2bf(x1.z); a[7]  = f2bf(x1.w);
                a[8]  = f2bf(y0.x); a[9]  = f2bf(y0.y); a[10] = f2bf(y0.z); a[11] = f2bf(y0.w);
                a[12] = f2bf(y1.x); a[13] = f2bf(y1.y); a[14] = f2bf(y1.z); a[15] = f2bf(y1.w);
            }
            a4[kk] = a;
        }
    };

    for (int tp = 0; tp < PAIRS_PER_WAVE; ++tp) {
        // Two consecutive 16-edge tiles; 32 consecutive edges per pair.
        const int ebase0 = blockBase + (wave * PAIRS_PER_WAVE + tp) * 32;

        // Graph id gather: all 32 lanes load one edge each (lanes 0-15 ->
        // tile 0, lanes 16-31 -> tile 1); invalid tail edges map to the dummy
        // row GRAPHS so the scatter below needs no divergent branch.
        int g16 = GRAPHS;
        {
            const int e = ebase0 + lane;
            if (e < E) {
                g16 = batch[edge_src[e]];
                atomicAdd(&scnt[g16], 1.0f);   // edge count per graph
            }
        }
        int grow0[8], grow1[8];
        #pragma unroll
        for (int r = 0; r < 8; ++r) {
            grow0[r] = __shfl(g16, r + 8 * lhalf, 32) * SSUM_STRIDE;
            grow1[r] = __shfl(g16, 16 + r + 8 * lhalf, 32) * SSUM_STRIDE;
        }

        // Two A-fragment sets (one per edge tile).
        v16bf afrag0[4], afrag1[4];
        load_afrags(ebase0 + l16, afrag0);
        load_afrags(ebase0 + 16 + l16, afrag1);

        // 8 N-tiles of 16 output columns; each B fragment (two 16B LDS reads
        // per lane: column n = nt*16 + lane%16, K {b..b+7, b+16..b+23}, high
        // lane-half +8 K) feeds BOTH edge tiles' WMMA chains.
        #pragma unroll 2
        for (int nt = 0; nt < 8; ++nt) {
            const int col = nt * 16 + l16;
            v8f acc0, acc1;
            #pragma unroll
            for (int i = 0; i < 8; ++i) { acc0[i] = bias[nt]; acc1[i] = bias[nt]; }

            #pragma unroll
            for (int kk = 0; kk < 4; ++kk) {
                const __bf16* bp = &w1t[col * W1T_STRIDE + kk * 32 + lhalf * 8];
                const v8bf blo = *reinterpret_cast<const v8bf*>(bp);
                const v8bf bhi = *reinterpret_cast<const v8bf*>(bp + 16);
                const v16bf bfrag = __builtin_shufflevector(
                    blo, bhi, 0, 1, 2, 3, 4, 5, 6, 7, 8, 9, 10, 11, 12, 13, 14, 15);
                acc0 = __builtin_amdgcn_wmma_f32_16x16x32_bf16(
                    false, afrag0[kk], false, bfrag, (short)0, acc0, false, false);
                acc1 = __builtin_amdgcn_wmma_f32_16x16x32_bf16(
                    false, afrag1[kk], false, bfrag, (short)0, acc1, false, false);
            }

            // ReLU + branch-free scatter into LDS accumulators.
            // D layout: VGPR r holds edge M = r + 8*lhalf, column N = lane%16.
            #pragma unroll
            for (int r = 0; r < 8; ++r) {
                atomicAdd(&ssum[grow0[r] + col], relu1(acc0[r]));
                atomicAdd(&ssum[grow1[r] + col], relu1(acc1[r]));
            }
        }
    }

    __syncthreads();
    // Flush per-block partials to global accumulators (dummy row dropped,
    // padding skipped).
    for (int i = tid; i < GRAPHS * HDIM; i += NTHREADS) {
        const int g = i >> 7, c = i & 127;
        atomicAdd(&gsums[i], ssum[g * SSUM_STRIDE + c]);
    }
    if (tid < GRAPHS)
        atomicAdd(&gcounts[tid], scnt[tid]);
}

// ---------------------------------------------------------------------------
// Kernel 2: mean + post-pool MLP (64x128 @ 128x128 -> relu -> @ 128x64).
// ~1.5 MFLOP total: one block of plain f32 is plenty.
// ---------------------------------------------------------------------------
__global__ __launch_bounds__(NTHREADS)
void finalize_kernel(const float* __restrict__ gsums,
                     const float* __restrict__ gcounts,
                     const float* __restrict__ W2,  // [128][128]
                     const float* __restrict__ b2,  // [128]
                     const float* __restrict__ W3,  // [128][64]
                     const float* __restrict__ b3,  // [64]
                     float*       __restrict__ out) {
    __shared__ float feat[GRAPHS * HDIM];
    __shared__ float t1[GRAPHS * HDIM];
    const int tid = threadIdx.x;

    for (int i = tid; i < GRAPHS * HDIM; i += NTHREADS) {
        const int g = i >> 7;
        float c = gcounts[g];
        c = c > 1.0f ? c : 1.0f;
        feat[i] = gsums[i] / c;
    }
    __syncthreads();

    for (int i = tid; i < GRAPHS * HDIM; i += NTHREADS) {
        const int g = i >> 7, c = i & 127;
        float acc = b2[c];
        #pragma unroll 8
        for (int k = 0; k < HDIM; ++k) acc += feat[g * HDIM + k] * W2[k * HDIM + c];
        t1[i] = acc > 0.0f ? acc : 0.0f;
    }
    __syncthreads();

    for (int i = tid; i < GRAPHS * OUTDIM; i += NTHREADS) {
        const int g = i >> 6, c = i & 63;
        float acc = b3[c];
        #pragma unroll 8
        for (int k = 0; k < HDIM; ++k) acc += t1[g * HDIM + k] * W3[k * OUTDIM + c];
        out[i] = acc;
    }
}

// ---------------------------------------------------------------------------
extern "C" void kernel_launch(void* const* d_in, const int* in_sizes, int n_in,
                              void* d_out, int out_size, void* d_ws, size_t ws_size,
                              hipStream_t stream) {
    const int*   edge_index = (const int*)  d_in[0];  // [2][E]
    const float* edge_attr  = (const float*)d_in[1];  // [E][128]
    const int*   batch      = (const int*)  d_in[2];  // [N]
    const float* W1         = (const float*)d_in[3];
    const float* b1         = (const float*)d_in[4];
    const float* W2         = (const float*)d_in[5];
    const float* b2         = (const float*)d_in[6];
    const float* W3         = (const float*)d_in[7];
    const float* b3         = (const float*)d_in[8];
    float* out = (float*)d_out;

    const int E = in_sizes[1] / INDIM;
    const int* edge_src = edge_index;  // row 0 of edge_index

    float* gsums   = (float*)d_ws;            // [64][128]
    float* gcounts = gsums + GRAPHS * HDIM;   // [64]

    zero_ws_kernel<<<1, NTHREADS, 0, stream>>>(gsums, GRAPHS * HDIM + GRAPHS);

    const int nblocks = (E + EPB - 1) / EPB;
    edge_mlp_scatter_kernel<<<nblocks, NTHREADS, 0, stream>>>(
        edge_src, edge_attr, batch, W1, b1, gsums, gcounts, E);

    finalize_kernel<<<1, NTHREADS, 0, stream>>>(gsums, gcounts, W2, b2, W3, b3, out);
}
